// DeepseekV32DecoderLayer_78237124263973
// MI455X (gfx1250) — matrix operand order repaired and checked by reading, not compiled, not run on hardware
//
#include <hip/hip_runtime.h>
#include <math.h>
#include <stdint.h>

// ---------------------------------------------------------------------------
// DeepSeek V3.2-style decoder layer for MI455X (gfx1250, wave32, WMMA).
// GEMMs: bf16 WMMA, fp32 accumulate.  B tiles staged to LDS with
// GLOBAL_LOAD_ASYNC_TO_LDS_B128 (double buffered, ASYNCcnt), converted to a
// bank-conflict-free bf16 LDS tile once per block; A-side activations are
// produced in bf16 by the upstream kernels, so the WMMA hot loop has zero
// fp32->bf16 conversions.
// ---------------------------------------------------------------------------

typedef __attribute__((ext_vector_type(16))) __bf16 v16bf;
typedef __attribute__((ext_vector_type(8)))  float  v8f;

#define T_    2048
#define D_    2048
#define H_    16
#define QL_   768
#define KVL_  512
#define NOPE_ 128
#define ROPE_ 64
#define VD_   128
#define IH_   16
#define ID_   128
#define TOPK_ 256
#define NE_   8
#define NF_   1024
#define FS_   1024

#define SCALE_F      0.07216878364870322f   // (NOPE+ROPE)^-0.5
#define IDX_SCALE_F  0.08838834764831845f   // ID^-0.5
#define RSF_F        2.5f
#define LOG10000_F   9.210340371976184f

#define BK_    64    // K-step staged per block
#define BPAD_  68    // fp32 staging row stride (floats): 4-bank rotation/row
#define BSTR_  72    // bf16 tile row stride (bf16): 36-bank rotation/row

// ---------------------------------------------------------------------------
// Generic GEMM: C[M,N] = A[M,K] @ B[N,K]^T   (A bf16, B fp32 [out,in], C fp32)
// Block = 256 threads = 8 waves; wave computes 16x64 C strip; block 128x64.
// Requires M % 128 == 0 and K % 64 == 0 (true for all call sites).
// ---------------------------------------------------------------------------
__global__ __launch_bounds__(256) void gemm_bf16_wmma(
    const __bf16* __restrict__ A, const float* __restrict__ B,
    float* __restrict__ C, int M, int N, int K) {
  __shared__ float  Bs32[2][64 * BPAD_];   // async-to-LDS fp32 staging
  __shared__ __bf16 Bs16[2][64 * BSTR_];   // bf16 fragment tile

  const int tid     = threadIdx.x;
  const int wave    = tid >> 5;
  const int lane    = tid & 31;
  const int lane_lo = lane & 15;
  const int hi      = lane >> 4;

  const int m0 = blockIdx.y * 128 + wave * 16;
  const int n0 = blockIdx.x * 64;

  // staging assignment: 64 rows x 64 floats; 4 threads/row, 4x b128 each
  const int srow = tid >> 2;          // 0..63
  const int scol = (tid & 3) * 16;    // 0,16,32,48
  int grow = n0 + srow; if (grow >= N) grow = N - 1;   // clamp (store guarded)
  const float* gB = B + (size_t)grow * K + scol;

  v8f zero = {0.f,0.f,0.f,0.f,0.f,0.f,0.f,0.f};
  v8f acc[4];
#pragma unroll
  for (int i = 0; i < 4; ++i) acc[i] = zero;

  const __bf16* Arow = A + (size_t)(m0 + lane_lo) * K;

  // issue async stage of tile 0 into buffer 0
  {
    unsigned l0 = (unsigned)(uintptr_t)&Bs32[0][srow * BPAD_ + scol];
#pragma unroll
    for (int j = 0; j < 4; ++j) {
      asm volatile("global_load_async_to_lds_b128 %0, %1, off"
                   :: "v"(l0 + (unsigned)(j * 16)), "v"(gB + j * 4) : "memory");
    }
  }

  for (int kt = 0; kt < K; kt += BK_) {
    const int buf = (kt >> 6) & 1;
    __syncthreads();   // prior readers of Bs32[buf^1]/Bs16[buf] are done
    if (kt + BK_ < K) {
      unsigned l0 = (unsigned)(uintptr_t)&Bs32[buf ^ 1][srow * BPAD_ + scol];
      const float* g = gB + kt + BK_;
#pragma unroll
      for (int j = 0; j < 4; ++j) {
        asm volatile("global_load_async_to_lds_b128 %0, %1, off"
                     :: "v"(l0 + (unsigned)(j * 16)), "v"(g + j * 4) : "memory");
      }
      // 4 ops of tile kt+BK_ in flight; tile kt's 4 (in-order) are complete
      asm volatile("s_wait_asynccnt 0x4" ::: "memory");
    } else {
      asm volatile("s_wait_asynccnt 0x0" ::: "memory");
    }

    // each thread converts exactly the fp32 cells it staged -> bf16 tile
    {
      const float* src = &Bs32[buf][srow * BPAD_ + scol];
      __bf16*      dst = &Bs16[buf][srow * BSTR_ + scol];
#pragma unroll
      for (int j = 0; j < 16; ++j) dst[j] = (__bf16)src[j];
    }
    __syncthreads();   // bf16 tile complete

#pragma unroll
    for (int ks = 0; ks < 2; ++ks) {
      v16bf a;
      const int ka = kt + ks * 32 + hi * 16;
#pragma unroll
      for (int i = 0; i < 8; ++i) a[i]     = Arow[ka + i];
#pragma unroll
      for (int i = 0; i < 8; ++i) a[8 + i] = Arow[ka + 8 + i];

#pragma unroll
      for (int nb = 0; nb < 4; ++nb) {
        const __bf16* brow = &Bs16[buf][(nb * 16 + lane_lo) * BSTR_ + ks * 32 + hi * 16];
        v16bf b;
#pragma unroll
        for (int i = 0; i < 16; ++i) b[i] = brow[i];
        acc[nb] = __builtin_amdgcn_wmma_f32_16x16x32_bf16(
            false, a, false, b, (short)0, acc[nb], false, false);
      }
    }
  }

#pragma unroll
  for (int nb = 0; nb < 4; ++nb) {
    int col = n0 + nb * 16 + lane_lo;
    if (col >= N) continue;
#pragma unroll
    for (int r = 0; r < 8; ++r) {
      int row = m0 + r + 8 * hi;
      C[(size_t)row * N + col] = acc[nb][r];
    }
  }
}

// ---------------------------------------------------------------------------
// Elementwise add
// ---------------------------------------------------------------------------
__global__ void add_kernel(const float* __restrict__ a, const float* __restrict__ b,
                           float* __restrict__ c, int n) {
  int i = blockIdx.x * blockDim.x + threadIdx.x;
  if (i < n) c[i] = a[i] + b[i];
}

// ---------------------------------------------------------------------------
// RMSNorm over rows (strided input slice), bf16 output (GEMM A operand)
// ---------------------------------------------------------------------------
__global__ __launch_bounds__(256) void rmsnorm_kernel(
    const float* __restrict__ x, int xoff, int xstride,
    const float* __restrict__ w, __bf16* __restrict__ y, int dim) {
  const int r = blockIdx.x;
  const float* xr = x + (size_t)r * xstride + xoff;
  __bf16* yr = y + (size_t)r * dim;
  __shared__ float red[256];
  float s = 0.f;
  for (int i = threadIdx.x; i < dim; i += 256) { float v = xr[i]; s += v * v; }
  red[threadIdx.x] = s; __syncthreads();
  for (int off = 128; off > 0; off >>= 1) {
    if (threadIdx.x < off) red[threadIdx.x] += red[threadIdx.x + off];
    __syncthreads();
  }
  float inv = rsqrtf(red[0] / (float)dim + 1e-6f);
  for (int i = threadIdx.x; i < dim; i += 256) yr[i] = (__bf16)(xr[i] * inv * w[i]);
}

// ---------------------------------------------------------------------------
// RoPE + index-k layernorm + index-w scaling, one block per token (64 thr).
// ---------------------------------------------------------------------------
__global__ __launch_bounds__(64) void rope_kernel(
    const int* __restrict__ positions, const float* __restrict__ qkv,
    float* __restrict__ q_full, float* __restrict__ iq,
    const float* __restrict__ index_k_raw,
    const float* __restrict__ kn_w, const float* __restrict__ kn_b,
    const float* __restrict__ iw_raw,
    float* __restrict__ k_pe_r, float* __restrict__ ik, float* __restrict__ iw) {
  const int t = blockIdx.x;
  const int tid = threadIdx.x;
  __shared__ float cs[32], sn[32], ln[ID_], red[64];

  float pos = (float)positions[t];
  if (tid < 32) {
    float inv = __expf(-((float)(2 * tid) / (float)ROPE_) * LOG10000_F);
    float f = pos * inv;
    cs[tid] = cosf(f); sn[tid] = sinf(f);
  }
  __syncthreads();

  for (int p = tid; p < H_ * 32; p += 64) {
    int h = p >> 5, i = p & 31;
    float* base = q_full + (size_t)t * (H_ * (NOPE_ + ROPE_)) + h * (NOPE_ + ROPE_) + NOPE_;
    float x1 = base[2 * i], x2 = base[2 * i + 1];
    base[2 * i]     = x1 * cs[i] - x2 * sn[i];
    base[2 * i + 1] = x2 * cs[i] + x1 * sn[i];
  }

  if (tid < 32) {
    const float* kp = qkv + (size_t)t * (QL_ + KVL_ + ROPE_) + QL_ + KVL_;
    float x1 = kp[2 * tid], x2 = kp[2 * tid + 1];
    k_pe_r[(size_t)t * ROPE_ + 2 * tid]     = x1 * cs[tid] - x2 * sn[tid];
    k_pe_r[(size_t)t * ROPE_ + 2 * tid + 1] = x2 * cs[tid] + x1 * sn[tid];
  }

  for (int p = tid; p < IH_ * 32; p += 64) {
    int h = p >> 5, i = p & 31;
    float* base = iq + (size_t)t * (IH_ * ID_) + h * ID_;
    float x1 = base[i], x2 = base[i + 32];
    base[i]      = x1 * cs[i] - x2 * sn[i];
    base[i + 32] = x2 * cs[i] + x1 * sn[i];
  }

  const float* xk = index_k_raw + (size_t)t * ID_;
  float s = 0.f;
  for (int i = tid; i < ID_; i += 64) s += xk[i];
  red[tid] = s; __syncthreads();
  for (int off = 32; off > 0; off >>= 1) {
    if (tid < off) red[tid] += red[tid + off];
    __syncthreads();
  }
  float mean = red[0] / (float)ID_;
  __syncthreads();
  float v = 0.f;
  for (int i = tid; i < ID_; i += 64) { float d = xk[i] - mean; v += d * d; }
  red[tid] = v; __syncthreads();
  for (int off = 32; off > 0; off >>= 1) {
    if (tid < off) red[tid] += red[tid + off];
    __syncthreads();
  }
  float inv_std = rsqrtf(red[0] / (float)ID_ + 1e-6f);
  for (int i = tid; i < ID_; i += 64)
    ln[i] = (xk[i] - mean) * inv_std * kn_w[i] + kn_b[i];
  __syncthreads();

  for (int i = tid; i < ID_; i += 64) {
    float o;
    if (i < 32)       o = ln[i] * cs[i] - ln[i + 32] * sn[i];
    else if (i < 64)  { int j = i - 32; o = ln[i] * cs[j] + ln[j] * sn[j]; }
    else              o = ln[i];
    ik[(size_t)t * ID_ + i] = o;
  }

  if (tid < IH_) iw[t * IH_ + tid] = iw_raw[t * IH_ + tid] * 0.25f; // IH^-0.5
}

// ---------------------------------------------------------------------------
// Indexer scores: iscore[t,s] = sum_h iw[t,h] * relu(IDX_SCALE * iq[t,h]·ik[s])
// One wave per 16x16 (t,s) tile; 16 heads x 4 k-steps = 64 WMMAs per tile.
// ---------------------------------------------------------------------------
__global__ __launch_bounds__(32) void iscore_kernel(
    const float* __restrict__ iq, const float* __restrict__ ik,
    const float* __restrict__ iw, float* __restrict__ iscore) {
  const int t0 = blockIdx.y * 16, s0 = blockIdx.x * 16;
  if (s0 > t0 + 15) return;   // uniform
  const int lane = threadIdx.x;
  const int lane_lo = lane & 15, hi = lane >> 4;

  float tile[8] = {0.f,0.f,0.f,0.f,0.f,0.f,0.f,0.f};
  v8f zero = {0.f,0.f,0.f,0.f,0.f,0.f,0.f,0.f};

  for (int h = 0; h < IH_; ++h) {
    v8f acc = zero;
    const float* Ar = iq + (size_t)(t0 + lane_lo) * (IH_ * ID_) + h * ID_;
    const float* Brow0 = ik + (size_t)(s0 + lane_lo) * ID_;
    for (int k = 0; k < ID_; k += 32) {
      v16bf a, b;
      const int k0 = k + hi * 16;
#pragma unroll
      for (int i = 0; i < 8; ++i) a[i]     = (__bf16)Ar[k0 + i];
#pragma unroll
      for (int i = 0; i < 8; ++i) a[8 + i] = (__bf16)Ar[k0 + 8 + i];
      const float* Br = Brow0 + k + hi * 16;
#pragma unroll
      for (int i = 0; i < 16; ++i) b[i] = (__bf16)Br[i];
      acc = __builtin_amdgcn_wmma_f32_16x16x32_bf16(
          false, a, false, b, (short)0, acc, false, false);
    }
#pragma unroll
    for (int r = 0; r < 8; ++r) {
      int row = t0 + r + 8 * hi;
      tile[r] += iw[row * IH_ + h] * fmaxf(acc[r] * IDX_SCALE_F, 0.f);
    }
  }
#pragma unroll
  for (int r = 0; r < 8; ++r)
    iscore[(size_t)(t0 + r + 8 * hi) * T_ + s0 + lane_lo] = tile[r];
}

// ---------------------------------------------------------------------------
// Top-k selection per row via threshold bisection + deterministic fill.
// ---------------------------------------------------------------------------
__global__ __launch_bounds__(256) void topk_kernel(
    const float* __restrict__ iscore, int* __restrict__ topk) {
  const int t = blockIdx.x;
  const int tid = threadIdx.x;
  const float* row = iscore + (size_t)t * T_;
  const int n = t + 1;  // causal-valid keys
  int* out = topk + (size_t)t * TOPK_;

  if (n <= TOPK_) {
    for (int j = tid; j < TOPK_; j += 256) out[j] = (j < n) ? j : -1;
    return;
  }

  __shared__ float fred[256];
  __shared__ int   ired[256];
  __shared__ float s_lo, s_hi;

  float mx = -1e30f, mn = 1e30f;
  for (int s = tid; s < n; s += 256) { float v = row[s]; mx = fmaxf(mx, v); mn = fminf(mn, v); }
  fred[tid] = mx; __syncthreads();
  for (int off = 128; off > 0; off >>= 1) {
    if (tid < off) fred[tid] = fmaxf(fred[tid], fred[tid + off]);
    __syncthreads();
  }
  if (tid == 0) s_hi = fred[0];
  __syncthreads();
  fred[tid] = mn; __syncthreads();
  for (int off = 128; off > 0; off >>= 1) {
    if (tid < off) fred[tid] = fminf(fred[tid], fred[tid + off]);
    __syncthreads();
  }
  if (tid == 0) s_lo = fred[0] - 1.0f;
  __syncthreads();

  for (int it = 0; it < 40; ++it) {
    float mid = 0.5f * (s_lo + s_hi);
    int c = 0;
    for (int s = tid; s < n; s += 256) if (row[s] > mid) ++c;
    ired[tid] = c; __syncthreads();
    for (int off = 128; off > 0; off >>= 1) {
      if (tid < off) ired[tid] += ired[tid + off];
      __syncthreads();
    }
    if (tid == 0) { if (ired[0] > TOPK_) s_lo = mid; else s_hi = mid; }
    __syncthreads();
  }

  if (tid == 0) {
    float tau = s_hi, lo = s_lo;
    int c = 0;
    for (int s = 0; s < n && c < TOPK_; ++s) if (row[s] > tau) out[c++] = s;
    for (int s = 0; s < n && c < TOPK_; ++s) {
      float v = row[s];
      if (v <= tau && v > lo) out[c++] = s;
    }
    for (; c < TOPK_; ++c) out[c] = -1;
  }
}

// ---------------------------------------------------------------------------
// Gathered attention over the <=256 selected keys per (token, head).
// bf16 output (feeds the W_o GEMM as A operand).
// ---------------------------------------------------------------------------
__global__ __launch_bounds__(512) void attn_kernel(
    const float* __restrict__ q_full, const float* __restrict__ kv_full,
    const float* __restrict__ k_pe_r, const int* __restrict__ topk,
    __bf16* __restrict__ attn) {
  const int t = blockIdx.x;
  const int h = threadIdx.x >> 5;
  const int lane = threadIdx.x & 31;

  const float* q = q_full + (size_t)t * (H_ * 192) + h * 192 + lane * 6;
  float qr[6];
#pragma unroll
  for (int i = 0; i < 6; ++i) qr[i] = q[i];

  float m = -1e30f, l = 0.f;
  float acc[4] = {0.f, 0.f, 0.f, 0.f};
  const int* sel = topk + (size_t)t * TOPK_;

  for (int j = 0; j < TOPK_; ++j) {
    int s = sel[j];
    if (s < 0) break;  // tail padding (uniform across wave)
    const float* kvb = kv_full + (size_t)s * (H_ * 256) + h * 256;
    float p = 0.f;
#pragma unroll
    for (int i = 0; i < 6; ++i) {
      int d = lane * 6 + i;
      float kv = (d < NOPE_) ? kvb[d] : k_pe_r[(size_t)s * ROPE_ + (d - NOPE_)];
      p += qr[i] * kv;
    }
    for (int off = 16; off > 0; off >>= 1) p += __shfl_xor(p, off, 32);
    float sc = p * SCALE_F;
    float nm = fmaxf(m, sc);
    float corr = __expf(m - nm);
    float pe = __expf(sc - nm);
    m = nm; l = l * corr + pe;
#pragma unroll
    for (int i = 0; i < 4; ++i)
      acc[i] = acc[i] * corr + pe * kvb[NOPE_ + lane + 32 * i];
  }
  float inv = 1.f / fmaxf(l, 1e-20f);
#pragma unroll
  for (int i = 0; i < 4; ++i)
    attn[(size_t)t * (H_ * VD_) + h * VD_ + lane + 32 * i] = (__bf16)(acc[i] * inv);
}

// ---------------------------------------------------------------------------
// MoE gate: softmax over 8 experts, top-2, renormalize → comb[t][8]
// ---------------------------------------------------------------------------
__global__ void gate_kernel(const float* __restrict__ glog, float* __restrict__ comb) {
  int t = blockIdx.x * blockDim.x + threadIdx.x;
  if (t >= T_) return;
  const float* g = glog + (size_t)t * NE_;
  float mx = -1e30f;
  for (int e = 0; e < NE_; ++e) mx = fmaxf(mx, g[e]);
  float ex[NE_];
  for (int e = 0; e < NE_; ++e) ex[e] = __expf(g[e] - mx);
  int i0 = 0;
  for (int e = 1; e < NE_; ++e) if (ex[e] > ex[i0]) i0 = e;
  int i1 = (i0 == 0) ? 1 : 0;
  for (int e = 0; e < NE_; ++e) if (e != i0 && ex[e] > ex[i1]) i1 = e;
  float v0 = ex[i0], v1 = ex[i1];
  float inv = 1.f / (v0 + v1);
  for (int e = 0; e < NE_; ++e) comb[(size_t)t * NE_ + e] = 0.f;
  comb[(size_t)t * NE_ + i0] = v0 * inv;
  comb[(size_t)t * NE_ + i1] = v1 * inv;
}

// ---------------------------------------------------------------------------
// y = silu(g) * u, bf16 output (feeds down-projection GEMM as A operand)
// ---------------------------------------------------------------------------
__global__ void silu_mul_kernel(const float* __restrict__ g, const float* __restrict__ u,
                                __bf16* __restrict__ y, int n) {
  int i = blockIdx.x * blockDim.x + threadIdx.x;
  if (i < n) { float x = g[i]; y[i] = (__bf16)((x / (1.f + __expf(-x))) * u[i]); }
}

// ---------------------------------------------------------------------------
// routed += comb[t,e] * eo[t,d]
// ---------------------------------------------------------------------------
__global__ void moe_accum_kernel(const float* __restrict__ eo, const float* __restrict__ comb,
                                 int e, float* __restrict__ routed) {
  int i = blockIdx.x * blockDim.x + threadIdx.x;
  if (i < T_ * D_) routed[i] += comb[(size_t)(i / D_) * NE_ + e] * eo[i];
}

// ---------------------------------------------------------------------------
// out = routed*RSF + shared
// ---------------------------------------------------------------------------
__global__ void final_kernel(const float* __restrict__ routed, const float* __restrict__ sh,
                             float* __restrict__ out, int n) {
  int i = blockIdx.x * blockDim.x + threadIdx.x;
  if (i < n) out[i] = routed[i] * RSF_F + sh[i];
}

// ---------------------------------------------------------------------------
// Host launch
// ---------------------------------------------------------------------------
extern "C" void kernel_launch(void* const* d_in, const int* in_sizes, int n_in,
                              void* d_out, int out_size, void* d_ws, size_t ws_size,
                              hipStream_t stream) {
  (void)in_sizes; (void)n_in; (void)out_size; (void)ws_size;

  const int*   positions = (const int*)  d_in[0];
  const float* hidden    = (const float*)d_in[1];
  const float* residual  = (const float*)d_in[2];
  const float* w_in      = (const float*)d_in[3];
  const float* w_post    = (const float*)d_in[4];
  const float* W_qkv_a   = (const float*)d_in[5];
  const float* w_q_norm  = (const float*)d_in[6];
  const float* W_qb      = (const float*)d_in[7];
  const float* w_kv_norm = (const float*)d_in[8];
  const float* W_kvb     = (const float*)d_in[9];
  const float* W_o       = (const float*)d_in[10];
  const float* W_idx_q   = (const float*)d_in[11];
  const float* W_idx_k   = (const float*)d_in[12];
  const float* idx_kn_w  = (const float*)d_in[13];
  const float* idx_kn_b  = (const float*)d_in[14];
  const float* W_idx_w   = (const float*)d_in[15];
  const float* W_gate    = (const float*)d_in[16];
  const float* Wg_exp    = (const float*)d_in[17];
  const float* Wu_exp    = (const float*)d_in[18];
  const float* Wd_exp    = (const float*)d_in[19];
  const float* Wg_sh     = (const float*)d_in[20];
  const float* Wu_sh     = (const float*)d_in[21];
  const float* Wd_sh     = (const float*)d_in[22];

  float* out    = (float*)d_out;                    // [0, T*D)   routed+shared
  float* res2   = (float*)d_out + (size_t)T_ * D_;  // [T*D, 2TD) residual

  float* w = (float*)d_ws;
  size_t off = 0;
  auto alloc  = [&](size_t n) { float* p = w + off; off += n; return p; };
  auto allocb = [&](size_t n) { __bf16* p = (__bf16*)(w + off); off += (n + 1) / 2; return p; };

  float*  res1     = alloc((size_t)T_ * D_);
  __bf16* h_b      = allocb((size_t)T_ * D_);
  float*  qkv      = alloc((size_t)T_ * (QL_ + KVL_ + ROPE_));
  float*  index_k  = alloc((size_t)T_ * ID_);
  float*  iw_raw   = alloc((size_t)T_ * IH_);
  __bf16* qcn_b    = allocb((size_t)T_ * QL_);
  __bf16* kvcn_b   = allocb((size_t)T_ * KVL_);
  float*  q_full   = alloc((size_t)T_ * H_ * (NOPE_ + ROPE_));
  float*  kv_full  = alloc((size_t)T_ * H_ * (NOPE_ + VD_));
  float*  iq       = alloc((size_t)T_ * IH_ * ID_);
  float*  ik       = alloc((size_t)T_ * ID_);
  float*  iw       = alloc((size_t)T_ * IH_);
  float*  k_pe_r   = alloc((size_t)T_ * ROPE_);
  float*  iscore   = alloc((size_t)T_ * T_);
  int*    topk     = (int*)alloc((size_t)T_ * TOPK_);
  __bf16* attn_b   = allocb((size_t)T_ * H_ * VD_);
  float*  h2       = alloc((size_t)T_ * D_);
  __bf16* h3_b     = allocb((size_t)T_ * D_);
  float*  glog     = alloc((size_t)T_ * NE_);
  float*  comb     = alloc((size_t)T_ * NE_);
  float*  routed   = alloc((size_t)T_ * D_);
  float*  gx       = alloc((size_t)T_ * NF_);
  float*  ux       = alloc((size_t)T_ * NF_);
  __bf16* gx_b     = allocb((size_t)T_ * NF_);
  float*  eo       = alloc((size_t)T_ * D_);

  auto gemm = [&](const __bf16* A, const float* B, float* C, int M, int N, int K) {
    dim3 grid((N + 63) / 64, (M + 127) / 128);
    gemm_bf16_wmma<<<grid, dim3(256), 0, stream>>>(A, B, C, M, N, K);
  };
  const int TD = T_ * D_;
  const int EW = 256;
  auto ew_grid = [&](int n) { return dim3((n + EW - 1) / EW); };

  // 1) residual add + input rmsnorm (bf16 activation for GEMMs)
  add_kernel<<<ew_grid(TD), EW, 0, stream>>>(hidden, residual, res1, TD);
  rmsnorm_kernel<<<T_, 256, 0, stream>>>(res1, 0, D_, w_in, h_b, D_);

  // 2) projections from h
  gemm(h_b, W_qkv_a, qkv,     T_, QL_ + KVL_ + ROPE_, D_);
  gemm(h_b, W_idx_k, index_k, T_, ID_,                D_);
  gemm(h_b, W_idx_w, iw_raw,  T_, IH_,                D_);

  // 3) low-rank norms + up-projections
  rmsnorm_kernel<<<T_, 256, 0, stream>>>(qkv, 0,   QL_ + KVL_ + ROPE_, w_q_norm,  qcn_b,  QL_);
  rmsnorm_kernel<<<T_, 256, 0, stream>>>(qkv, QL_, QL_ + KVL_ + ROPE_, w_kv_norm, kvcn_b, KVL_);
  gemm(qcn_b,  W_qb,    q_full,  T_, H_ * (NOPE_ + ROPE_), QL_);
  gemm(kvcn_b, W_kvb,   kv_full, T_, H_ * (NOPE_ + VD_),   KVL_);
  gemm(qcn_b,  W_idx_q, iq,      T_, IH_ * ID_,            QL_);

  // 4) RoPE / layernorm / index weights
  rope_kernel<<<T_, 64, 0, stream>>>(positions, qkv, q_full, iq, index_k,
                                     idx_kn_w, idx_kn_b, iw_raw, k_pe_r, ik, iw);

  // 5) indexer scores (WMMA), top-k selection
  {
    dim3 grid(T_ / 16, T_ / 16);
    iscore_kernel<<<grid, 32, 0, stream>>>(iq, ik, iw, iscore);
  }
  topk_kernel<<<T_, 256, 0, stream>>>(iscore, topk);

  // 6) gathered sparse attention (bf16 output)
  attn_kernel<<<T_, 512, 0, stream>>>(q_full, kv_full, k_pe_r, topk, attn_b);

  // 7) output projection, residual, post norm
  gemm(attn_b, W_o, h2, T_, D_, H_ * VD_);
  add_kernel<<<ew_grid(TD), EW, 0, stream>>>(h2, res1, res2, TD);
  rmsnorm_kernel<<<T_, 256, 0, stream>>>(res2, 0, D_, w_post, h3_b, D_);

  // 8) MoE routing
  gemm(h3_b, W_gate, glog, T_, NE_, D_);
  gate_kernel<<<(T_ + 255) / 256, 256, 0, stream>>>(glog, comb);
  hipMemsetAsync(routed, 0, (size_t)TD * sizeof(float), stream);

  // 9) dense experts (matches reference dense einsum)
  for (int e = 0; e < NE_; ++e) {
    const float* Wg = Wg_exp + (size_t)e * NF_ * D_;
    const float* Wu = Wu_exp + (size_t)e * NF_ * D_;
    const float* Wd = Wd_exp + (size_t)e * D_ * NF_;
    gemm(h3_b, Wg, gx, T_, NF_, D_);
    gemm(h3_b, Wu, ux, T_, NF_, D_);
    silu_mul_kernel<<<ew_grid(T_ * NF_), EW, 0, stream>>>(gx, ux, gx_b, T_ * NF_);
    gemm(gx_b, Wd, eo, T_, D_, NF_);
    moe_accum_kernel<<<ew_grid(TD), EW, 0, stream>>>(eo, comb, e, routed);
  }

  // 10) shared expert + final combine
  gemm(h3_b, Wg_sh, gx, T_, FS_, D_);
  gemm(h3_b, Wu_sh, ux, T_, FS_, D_);
  silu_mul_kernel<<<ew_grid(T_ * FS_), EW, 0, stream>>>(gx, ux, gx_b, T_ * FS_);
  gemm(gx_b, Wd_sh, eo, T_, D_, FS_);
  final_kernel<<<ew_grid(TD), EW, 0, stream>>>(routed, eo, out, TD);
}
